// CausalSelfAttention_73744588472705
// MI455X (gfx1250) — compile-verified, matrix operand order checked by reading, not run on hardware
//
#include <hip/hip_runtime.h>
#include <math.h>

// ---------------------------------------------------------------------------
// CausalSelfAttention with memory prefix (ssmax), MI455X / gfx1250.
// cvt(f32->bf16) -> QKV GEMM (wmma bf16, fold logT*qm/sqrt(dh) into Q)
//                -> flash attention (wmma scores + online softmax + wmma P*V)
//                -> output projection GEMM (wmma bf16, f32 out).
// Matrix math: v_wmma_f32_16x16x32_bf16, f32 accumulate.
// Tile staging: TDM tensor_load_to_lds (TENSORcnt) where no transpose is
// needed; manual transpose staging for V; global_prefetch for next tiles.
// ---------------------------------------------------------------------------

typedef __attribute__((ext_vector_type(16))) __bf16 bf16x16;
typedef __attribute__((ext_vector_type(8)))  __bf16 bf16x8;
typedef __attribute__((ext_vector_type(8)))  float  f32x8;

#define B_SZ   2
#define T_SZ   2048
#define C_SZ   1024
#define H_SZ   16
#define DH     64
#define C3     3072
#define MEMP   1024

#if defined(__gfx1250__) && __has_builtin(__builtin_amdgcn_tensor_load_to_lds)
#define USE_TDM 1
#else
#define USE_TDM 0
#endif

#if USE_TDM
typedef __attribute__((ext_vector_type(4))) unsigned int u32x4;
typedef __attribute__((ext_vector_type(8))) int          i32x8;
typedef __attribute__((ext_vector_type(4))) int          i32x4;

// ---------------------------------------------------------------------------
// Issue a TDM 2D tile load: tile_h rows x tile_w bf16 elements, source row
// stride stride_elems, packed contiguously into LDS at lds_byte.
// D# layout per CDNA5 ISA ch.8: group0 = {count=1, lds_addr, global_addr,
// type=2}; group1 = {data_size=2B, tensor_dim0=stride, tensor_dim1=huge,
// tile_dim0/1, tensor_dim0_stride}. Groups 2/3 zero (2D tile).
// Issue from ONE wave; EXEC is ignored; tracked on TENSORcnt.
// ---------------------------------------------------------------------------
__device__ __forceinline__ void tdm_load_2d(const void* g, unsigned lds_byte,
                                            unsigned tile_w, unsigned tile_h,
                                            unsigned stride_elems) {
  unsigned long long ga = (unsigned long long)(size_t)g;
  unsigned dim0 = stride_elems;   // tensor width (elements) >= tile_w
  unsigned dim1 = 1u << 30;       // effectively unbounded row count
  u32x4 g0;
  g0[0] = 1u;                                              // count=1
  g0[1] = lds_byte;                                        // lds_addr
  g0[2] = (unsigned)ga;                                    // global_addr lo
  g0[3] = (unsigned)((ga >> 32) & 0x01FFFFFFull) | (2u << 30);  // hi | type=2
  i32x8 g1;
  g1[0] = (int)(1u << 16);                                 // data_size = 2B
  g1[1] = (int)((dim0 & 0xFFFFu) << 16);                   // dim0 lo16 @63:48
  g1[2] = (int)(((dim0 >> 16) & 0xFFFFu) | ((dim1 & 0xFFFFu) << 16));
  g1[3] = (int)(((dim1 >> 16) & 0xFFFFu) | (tile_w << 16));// dim1 hi | tile0
  g1[4] = (int)(tile_h & 0xFFFFu);                         // tile_dim1
  g1[5] = (int)stride_elems;                               // dim0_stride lo32
  g1[6] = 0;
  g1[7] = 0;
  i32x4 z4 = {0, 0, 0, 0};
#if __clang_major__ >= 23
  i32x8 z8 = {0, 0, 0, 0, 0, 0, 0, 0};
  __builtin_amdgcn_tensor_load_to_lds(g0, g1, z4, z4, z8, 0);
#else
  __builtin_amdgcn_tensor_load_to_lds(g0, g1, z4, z4, 0);
#endif
}
#endif  // USE_TDM

// ---------------------------------------------------------------------------
// Fragment loader for 16x32 bf16 tiles (A-layout per ISA 7.12.2; also used for
// B operands since B columns are stored as contiguous rows in our staging).
// lane 0-15 : row = lane,    K in {0..7, 16..23}
// lane 16-31: row = lane-16, K in {8..15, 24..31}
// Requires: base 16B aligned, ld multiple of 8 elements.
// ---------------------------------------------------------------------------
__device__ __forceinline__ bf16x16 load_frag(const __bf16* base, int ld) {
  int lane = threadIdx.x & 31;
  const __bf16* p = base + (lane & 15) * ld + ((lane >> 4) << 3);
  bf16x8 lo = *(const bf16x8*)(p);
  bf16x8 hi = *(const bf16x8*)(p + 16);
  return __builtin_shufflevector(lo, hi, 0, 1, 2, 3, 4, 5, 6, 7,
                                         8, 9, 10, 11, 12, 13, 14, 15);
}

__device__ __forceinline__ f32x8 wmma_bf16(bf16x16 a, bf16x16 b, f32x8 c) {
  // (neg_a, A, neg_b, B, c_mod, C, reuse_a, reuse_b)
  return __builtin_amdgcn_wmma_f32_16x16x32_bf16(false, a, false, b,
                                                 (short)0, c, false, false);
}

// ---------------------------------------------------------------------------
// fp32 -> bf16 conversion
// ---------------------------------------------------------------------------
__global__ void cvt_f32_bf16(const float* __restrict__ src,
                             __bf16* __restrict__ dst, int n) {
  int i = blockIdx.x * blockDim.x + threadIdx.x;
  if (i < n) dst[i] = (__bf16)src[i];
}

// ---------------------------------------------------------------------------
// GEMM: out[M,N] = A[M,K] @ W[N,K]^T      (bf16 in, f32 acc)
// If outb != null: write bf16, scaling columns gN < Cq by logT*qm[gN%64]/8.
// Else: write f32 to outf.
// Block: 256 threads (8 waves). Tile 64(M) x 128(N), K step 32 via LDS.
// A/B tiles staged by the Tensor Data Mover (no transpose needed).
// ---------------------------------------------------------------------------
__global__ __launch_bounds__(256)
void gemm_bf16_kernel(const __bf16* __restrict__ A,
                      const __bf16* __restrict__ W,
                      __bf16* __restrict__ outb,
                      float* __restrict__ outf,
                      const float* __restrict__ qm,
                      int M, int N, int K, float logT, int Cq) {
  __shared__ __bf16 ldsA[64 * 32];    // 4 KB
  __shared__ __bf16 ldsB[128 * 32];   // 8 KB

  const int mBase = blockIdx.y * 64;
  const int nBase = blockIdx.x * 128;
  const int wave = threadIdx.x >> 5;
  const int lane = threadIdx.x & 31;
  const int mSub = (wave & 3) * 16;
  const int nSub = (wave >> 2) * 64;

  f32x8 acc[4];
#pragma unroll
  for (int c = 0; c < 4; ++c) acc[c] = (f32x8){0.f, 0.f, 0.f, 0.f, 0.f, 0.f, 0.f, 0.f};

  for (int k0 = 0; k0 < K; k0 += 32) {
    __syncthreads();
#if USE_TDM
    if (wave == 0) {
      tdm_load_2d(A + (size_t)mBase * K + k0, (unsigned)(size_t)ldsA, 32, 64,
                  (unsigned)K);
      tdm_load_2d(W + (size_t)nBase * K + k0, (unsigned)(size_t)ldsB, 32, 128,
                  (unsigned)K);
      __builtin_amdgcn_s_wait_tensorcnt(0);
    }
#else
    // Stage A: 64 rows x 32 k = 256 chunks of 8
    {
      int row = threadIdx.x >> 2;
      int kc = (threadIdx.x & 3) << 3;
      const __bf16* srcA = A + (size_t)(mBase + row) * K + k0 + kc;
      *(bf16x8*)(ldsA + row * 32 + kc) = *(const bf16x8*)srcA;
      if (k0 + 32 < K) __builtin_prefetch(srcA + 32, 0, 1);
    }
    // Stage B: 128 rows x 32 k = 512 chunks of 8
#pragma unroll
    for (int i = 0; i < 2; ++i) {
      int cidx = threadIdx.x + i * 256;
      int row = cidx >> 2;
      int kc = (cidx & 3) << 3;
      const __bf16* srcB = W + (size_t)(nBase + row) * K + k0 + kc;
      *(bf16x8*)(ldsB + row * 32 + kc) = *(const bf16x8*)srcB;
      if (k0 + 32 < K) __builtin_prefetch(srcB + 32, 0, 1);
    }
#endif
    __syncthreads();

    bf16x16 af = load_frag(ldsA + mSub * 32, 32);
#pragma unroll
    for (int c = 0; c < 4; ++c) {
      bf16x16 bf = load_frag(ldsB + (nSub + c * 16) * 32, 32);
      acc[c] = wmma_bf16(af, bf, acc[c]);
    }
  }

  // Writeback (C-layout: element r -> row r + 8*(lane>>4), col lane&15)
  const int col0 = lane & 15;
  const int rh = (lane >> 4) << 3;
#pragma unroll
  for (int c = 0; c < 4; ++c) {
    int gN = nBase + nSub + c * 16 + col0;
    float scale = 1.0f;
    if (outb != nullptr && qm != nullptr && gN < Cq)
      scale = logT * qm[gN & (DH - 1)] * 0.125f;  // 1/sqrt(64) = 0.125
#pragma unroll
    for (int r = 0; r < 8; ++r) {
      int gM = mBase + mSub + rh + r;
      float v = acc[c][r] * scale;
      if (outb != nullptr)
        outb[(size_t)gM * N + gN] = (__bf16)v;
      else
        outf[(size_t)gM * N + gN] = v;
    }
  }
}

// ---------------------------------------------------------------------------
// Flash attention with memory-prefix causal mask.
// Grid: B*H*(T/64) blocks, 128 threads (4 waves). Wave w owns 16 query rows.
// Per 32-key step: K tile (32x64) staged by TDM; V^T tile (64x32) staged
// manually (transpose); scores via 4 WMMAs, analytic mask (j<=i || j<MEM),
// online softmax (shfl_xor over 16-lane halves), P->LDS->A-frag, 4 WMMAs P*V.
// ---------------------------------------------------------------------------
__global__ __launch_bounds__(128)
void attn_kernel(const __bf16* __restrict__ qkv,  // [B*T, 3C] bf16 (Q prescaled)
                 __bf16* __restrict__ y) {        // [B*T, C] bf16
  __shared__ __bf16 ldsK[32 * 64];       // 4 KB  (key-major, feat contiguous)
  __shared__ __bf16 ldsVt[64 * 32];      // 4 KB  (feat-major, key contiguous)
  __shared__ __bf16 ldsP[4 * 16 * 32];   // 4 KB  (per-wave P tiles)

  const int qb = blockIdx.x & 31;           // T/64 = 32 query blocks
  const int h = (blockIdx.x >> 5) & 15;
  const int b = blockIdx.x >> 9;
  const int wave = threadIdx.x >> 5;
  const int lane = threadIdx.x & 31;
  const int qBase = qb * 64 + wave * 16;
  const int col0 = lane & 15;
  const int rh = (lane >> 4) << 3;

  // Q fragments (feat 0..31 and 32..63), loaded once.
  const __bf16* qptr = qkv + ((size_t)(b * T_SZ + qBase)) * C3 + h * DH;
  bf16x16 qf0 = load_frag(qptr, C3);
  bf16x16 qf1 = load_frag(qptr + 32, C3);

  const float NEGB = -3.0e38f;
  float mrow[8], lrow[8];
  f32x8 oacc[4];
#pragma unroll
  for (int r = 0; r < 8; ++r) { mrow[r] = NEGB; lrow[r] = 0.f; }
#pragma unroll
  for (int c = 0; c < 4; ++c) oacc[c] = (f32x8){0.f, 0.f, 0.f, 0.f, 0.f, 0.f, 0.f, 0.f};

  int kEnd = qb * 64 + 64;
  if (kEnd < MEMP) kEnd = MEMP;   // memory prefix always visible

  for (int kt = 0; kt < kEnd; kt += 32) {
    __syncthreads();  // previous K/V/P consumers done
#if USE_TDM
    if (wave == 0)
      tdm_load_2d(qkv + ((size_t)(b * T_SZ + kt)) * C3 + C_SZ + h * DH,
                  (unsigned)(size_t)ldsK, 64, 32, C3);
#endif
    // Stage V^T (64 feats x 32 keys) manually (needs transpose); K too if no TDM.
#pragma unroll
    for (int i = 0; i < 2; ++i) {
      int cid = threadIdx.x + i * 128;        // 0..255 chunks of 8
      int key = cid >> 3;
      int f0 = (cid & 7) << 3;
      const __bf16* src =
          qkv + ((size_t)(b * T_SZ + kt + key)) * C3 + C_SZ + h * DH + f0;
#if !USE_TDM
      *(bf16x8*)(ldsK + key * 64 + f0) = *(const bf16x8*)src;
#endif
      bf16x8 vv = *(const bf16x8*)(src + C_SZ);   // V at +C offset
#pragma unroll
      for (int j = 0; j < 8; ++j) ldsVt[(f0 + j) * 32 + key] = vv[j];
      if (kt + 32 < kEnd) {                        // prefetch next key tile
        __builtin_prefetch(src + (size_t)32 * C3, 0, 1);          // next K
        __builtin_prefetch(src + (size_t)32 * C3 + C_SZ, 0, 1);   // next V
      }
    }
#if USE_TDM
    if (wave == 0) __builtin_amdgcn_s_wait_tensorcnt(0);
#endif
    __syncthreads();

    // Scores: S0 = Q x K[0:16]^T, S1 = Q x K[16:32]^T   (dh=64 -> 2 k-steps)
    f32x8 s0 = (f32x8){0.f, 0.f, 0.f, 0.f, 0.f, 0.f, 0.f, 0.f};
    f32x8 s1 = s0;
    s0 = wmma_bf16(qf0, load_frag(ldsK, 64), s0);
    s0 = wmma_bf16(qf1, load_frag(ldsK + 32, 64), s0);
    s1 = wmma_bf16(qf0, load_frag(ldsK + 16 * 64, 64), s1);
    s1 = wmma_bf16(qf1, load_frag(ldsK + 16 * 64 + 32, 64), s1);

    // Mask: attend iff (tk <= tq) || (tk < MEM)
    const int tk0 = kt + col0;
    const int tk1 = tk0 + 16;
    float al8[8];
#pragma unroll
    for (int r = 0; r < 8; ++r) {
      int tq = qBase + rh + r;
      if (!((tk0 <= tq) || (tk0 < MEMP))) s0[r] = NEGB;
      if (!((tk1 <= tq) || (tk1 < MEMP))) s1[r] = NEGB;

      // Online softmax (row reduction across the 16 lanes of this half-wave)
      float v = fmaxf(s0[r], s1[r]);
      v = fmaxf(v, __shfl_xor(v, 1, 32));
      v = fmaxf(v, __shfl_xor(v, 2, 32));
      v = fmaxf(v, __shfl_xor(v, 4, 32));
      v = fmaxf(v, __shfl_xor(v, 8, 32));
      float mn = fmaxf(mrow[r], v);
      float al = __expf(mrow[r] - mn);
      float p0 = __expf(s0[r] - mn);
      float p1 = __expf(s1[r] - mn);
      float sm = p0 + p1;
      sm += __shfl_xor(sm, 1, 32);
      sm += __shfl_xor(sm, 2, 32);
      sm += __shfl_xor(sm, 4, 32);
      sm += __shfl_xor(sm, 8, 32);
      lrow[r] = lrow[r] * al + sm;
      mrow[r] = mn;
      al8[r] = al;
      s0[r] = p0;
      s1[r] = p1;
    }
#pragma unroll
    for (int c = 0; c < 4; ++c)
#pragma unroll
      for (int r = 0; r < 8; ++r) oacc[c][r] *= al8[r];

    // P (16x32) -> per-wave LDS (C-layout store, A-layout reload)
    __bf16* pw = ldsP + wave * 512;
#pragma unroll
    for (int r = 0; r < 8; ++r) {
      pw[(rh + r) * 32 + col0] = (__bf16)s0[r];
      pw[(rh + r) * 32 + 16 + col0] = (__bf16)s1[r];
    }
    // O += P x V   (4 column tiles of 16 over dh=64)
    bf16x16 pf = load_frag(pw, 32);
#pragma unroll
    for (int c = 0; c < 4; ++c)
      oacc[c] = wmma_bf16(pf, load_frag(ldsVt + c * 16 * 32, 32), oacc[c]);
  }

  // Normalize and store Y[b, t, h*dh + n]
#pragma unroll
  for (int c = 0; c < 4; ++c) {
    int gcol = h * DH + c * 16 + col0;
#pragma unroll
    for (int r = 0; r < 8; ++r) {
      int t = qBase + rh + r;
      float val = oacc[c][r] / lrow[r];
      y[((size_t)(b * T_SZ + t)) * C_SZ + gcol] = (__bf16)val;
    }
  }
}

// ---------------------------------------------------------------------------
// Launch
// ---------------------------------------------------------------------------
extern "C" void kernel_launch(void* const* d_in, const int* in_sizes, int n_in,
                              void* d_out, int out_size, void* d_ws, size_t ws_size,
                              hipStream_t stream) {
  const float* x      = (const float*)d_in[0];  // [B,T,C]
  const float* w_qkv  = (const float*)d_in[1];  // [3C,C]
  const float* w_proj = (const float*)d_in[2];  // [C,C]
  const float* qm     = (const float*)d_in[3];  // [dh]
  // d_in[4] attn_mask: structurally (j<=i)|(j<MEM) -> computed analytically.
  float* out = (float*)d_out;

  char* ws = (char*)d_ws;
  __bf16* xb   = (__bf16*)(ws);                        //  8 MB: x bf16
  __bf16* wqb  = (__bf16*)(ws + ((size_t)8 << 20));    //  6 MB: w_qkv bf16
  __bf16* wpb  = (__bf16*)(ws + ((size_t)14 << 20));   //  2 MB: w_proj bf16
  __bf16* qkvb = (__bf16*)(ws + ((size_t)16 << 20));   // 24 MB: QKV bf16
  __bf16* yb   = (__bf16*)(ws + ((size_t)40 << 20));   //  8 MB: attn out bf16

  const int M = B_SZ * T_SZ;          // 4096
  const int nx = M * C_SZ;            // 4,194,304
  const int nq = C3 * C_SZ;           // 3,145,728
  const int np = C_SZ * C_SZ;         // 1,048,576

  cvt_f32_bf16<<<(nx + 255) / 256, 256, 0, stream>>>(x, xb, nx);
  cvt_f32_bf16<<<(nq + 255) / 256, 256, 0, stream>>>(w_qkv, wqb, nq);
  cvt_f32_bf16<<<(np + 255) / 256, 256, 0, stream>>>(w_proj, wpb, np);

  // QKV projection (+ fold logT*qm/sqrt(dh) into Q columns [0,C))
  {
    dim3 grid(C3 / 128, M / 64);
    gemm_bf16_kernel<<<grid, 256, 0, stream>>>(
        xb, wqb, qkvb, nullptr, qm, M, C3, C_SZ, logf((float)T_SZ), C_SZ);
  }

  // Flash attention: B*H*(T/64) = 1024 blocks
  attn_kernel<<<B_SZ * H_SZ * (T_SZ / 64), 128, 0, stream>>>(qkvb, yb);

  // Output projection (f32 out)
  {
    dim3 grid(C_SZ / 128, M / 64);
    gemm_bf16_kernel<<<grid, 256, 0, stream>>>(
        yb, wpb, nullptr, out, nullptr, M, C_SZ, C_SZ, 0.f, 0);
  }
}
